// ReachabilityGNNv11_61632780697763
// MI455X (gfx1250) — matrix operand order repaired and checked by reading, not compiled, not run on hardware
//
#include <hip/hip_runtime.h>
#include <hip/hip_bf16.h>

typedef _Float16 h8  __attribute__((ext_vector_type(8)));
typedef _Float16 h16 __attribute__((ext_vector_type(16)));
typedef float    f8  __attribute__((ext_vector_type(8)));

// Native, non-returning f32 atomic add (device scope, relaxed) ->
// lowers to global_atomic_add_f32 regardless of -munsafe-fp-atomics.
__device__ __forceinline__ void atomic_fadd(float* p, float v) {
  __hip_atomic_fetch_add(p, v, __ATOMIC_RELAXED, __HIP_MEMORY_SCOPE_AGENT);
}

// ---------- float <-> monotonic uint key (for atomic segment-max) ----------
__device__ __forceinline__ unsigned f2key(float x) {
  unsigned u = __float_as_uint(x);
  return (u & 0x80000000u) ? ~u : (u | 0x80000000u);
}
__device__ __forceinline__ float key2f(unsigned k) {
  return __uint_as_float((k & 0x80000000u) ? (k ^ 0x80000000u) : ~k);
}

__device__ __forceinline__ void edge_sd(const int* __restrict__ ei, int t, int E,
                                        int& s, int& d) {
  if (t < E) { s = ei[t]; d = ei[E + t]; } else { s = t - E; d = t - E; }
}

// ---------- pack W (K x Nout, f32 row-major) into WMMA B-operand f16 layout ----
// B 32x16 f16 per (kstep s, ntile nt): lane L holds column n = L%16,
// K rows kb..kb+15 with kb = (L<16 ? 0 : 16), as 16 contiguous halves.
__global__ void pack_b_kernel(const float* __restrict__ W, _Float16* __restrict__ dst,
                              int Ksteps, int Ntiles, int Nout) {
  int idx = blockIdx.x * blockDim.x + threadIdx.x;
  int total = Ksteps * Ntiles * 512;
  if (idx >= total) return;
  int i    = idx & 15;
  int lane = (idx >> 4) & 31;
  int g    = idx >> 9;
  int nt   = g % Ntiles;
  int s    = g / Ntiles;
  int kb   = (lane < 16) ? 0 : 16;
  int k    = 32 * s + kb + i;
  int col  = nt * 16 + (lane & 15);
  dst[idx] = (_Float16)W[(size_t)k * Nout + col];
}

// ---------- climber embed: ce = relu(climber @ Wc + bc), stored f16 ----------
__global__ void climber_kernel(const float* __restrict__ climber,
                               const float* __restrict__ Wc,
                               const float* __restrict__ bc,
                               _Float16* __restrict__ ce16, int G) {
  int idx = blockIdx.x * blockDim.x + threadIdx.x;
  if (idx >= G * 64) return;
  int g = idx >> 6, k = idx & 63;
  float acc = bc[k];
#pragma unroll
  for (int j = 0; j < 4; j++) acc += climber[g * 4 + j] * Wc[j * 64 + k];
  ce16[idx] = (_Float16)fmaxf(acc, 0.f);
}

// ---------- layer-1 transform: hpre = x @ W1  (K=8, scalar, trivially cheap) ----
__global__ void gemm1_kernel(const float* __restrict__ x, const float* __restrict__ W1,
                             float* __restrict__ hpre, int nN) {
  int t = blockIdx.x * blockDim.x + threadIdx.x;
  if (t >= nN * 32) return;
  int n = t >> 5, q = t & 31;
  const float* xr = x + (size_t)n * 8;
  float xv[8];
#pragma unroll
  for (int k = 0; k < 8; k++) xv[k] = xr[k];
  float4 acc = make_float4(0.f, 0.f, 0.f, 0.f);
#pragma unroll
  for (int k = 0; k < 8; k++) {
    float4 w = *(const float4*)(W1 + k * 128 + q * 4);
    acc.x += xv[k] * w.x; acc.y += xv[k] * w.y;
    acc.z += xv[k] * w.z; acc.w += xv[k] * w.w;
  }
  *(float4*)(hpre + (size_t)n * 128 + q * 4) = acc;
}

// ---------- per-node attention terms + per-layer init of max-keys / denom ------
__global__ void alpha_init_kernel(const float* __restrict__ hpre,
                                  const float* __restrict__ a_src,
                                  const float* __restrict__ a_dst,
                                  float* __restrict__ als, float* __restrict__ ald,
                                  unsigned* __restrict__ emaxkey,
                                  float* __restrict__ denom, int nN) {
  int idx = blockIdx.x * blockDim.x + threadIdx.x;
  if (idx >= nN * 2) return;
  int n = idx >> 1, hh = idx & 1;
  const float* hr = hpre + (size_t)n * 128 + hh * 64;
  const float* as = a_src + hh * 64;
  const float* ad = a_dst + hh * 64;
  float ss = 0.f, dd = 0.f;
#pragma unroll 8
  for (int k = 0; k < 64; k++) { float v = hr[k]; ss += v * as[k]; dd += v * ad[k]; }
  als[idx] = ss;
  ald[idx] = dd;
  emaxkey[idx] = 0x007FFFFFu;  // key(-inf)
  denom[idx] = 0.f;
}

// ---------- edge pass 1: segment max of leaky-relu(e) over dst ----------------
__global__ void edge_max_kernel(const float* __restrict__ als,
                                const float* __restrict__ ald,
                                const int* __restrict__ ei,
                                unsigned* __restrict__ emaxkey, int E, int EN) {
  int t = blockIdx.x * blockDim.x + threadIdx.x;
  if (t >= EN) return;
  int s, d; edge_sd(ei, t, E, s, d);
#pragma unroll
  for (int hh = 0; hh < 2; hh++) {
    float e = als[s * 2 + hh] + ald[d * 2 + hh];
    e = (e > 0.f) ? e : 0.2f * e;
    atomicMax(&emaxkey[d * 2 + hh], f2key(e));
  }
}

// ---------- edge pass 2: ex = exp(e - emax[dst]); denom[dst] += ex ------------
__global__ void edge_exp_kernel(const float* __restrict__ als,
                                const float* __restrict__ ald,
                                const int* __restrict__ ei,
                                const unsigned* __restrict__ emaxkey,
                                float* __restrict__ denom,
                                float* __restrict__ exbuf, int E, int EN) {
  int t = blockIdx.x * blockDim.x + threadIdx.x;
  if (t >= EN) return;
  int s, d; edge_sd(ei, t, E, s, d);
#pragma unroll
  for (int hh = 0; hh < 2; hh++) {
    float e = als[s * 2 + hh] + ald[d * 2 + hh];
    e = (e > 0.f) ? e : 0.2f * e;
    float em = key2f(emaxkey[d * 2 + hh]);
    float ex = __expf(e - em);
    exbuf[(size_t)t * 2 + hh] = ex;
    atomic_fadd(&denom[d * 2 + hh], ex);
  }
}

// ---------- edge pass 3: agg[dst] += h[src] * alpha  (one wave32 per edge) ----
__global__ void edge_msg_kernel(const float* __restrict__ hpre,
                                const float* __restrict__ exbuf,
                                const float* __restrict__ denom,
                                const int* __restrict__ ei,
                                float* __restrict__ agg, int E, int EN) {
  int wid = (blockIdx.x * blockDim.x + threadIdx.x) >> 5;
  int lane = threadIdx.x & 31;
  if (wid >= EN) return;
  int s, d; edge_sd(ei, wid, E, s, d);
  int c0 = lane * 4;           // lanes 0-15 -> head 0 (cols 0..63), 16-31 -> head 1
  int hh = lane >> 4;
  float alpha = exbuf[(size_t)wid * 2 + hh] / (denom[d * 2 + hh] + 1e-16f);
  float4 hv = *(const float4*)(hpre + (size_t)s * 128 + c0);
  float* ap = agg + (size_t)d * 128 + c0;
  atomic_fadd(ap + 0, hv.x * alpha);
  atomic_fadd(ap + 1, hv.y * alpha);
  atomic_fadd(ap + 2, hv.z * alpha);
  atomic_fadd(ap + 3, hv.w * alpha);
}

// ---------- finalize: h = relu(mean_heads(agg) + b), write f16 ----------------
__global__ void finalize_kernel(const float* __restrict__ agg,
                                const float* __restrict__ bias,
                                _Float16* __restrict__ h16p, int nN) {
  int idx = blockIdx.x * blockDim.x + threadIdx.x;
  if (idx >= nN * 64) return;
  int n = idx >> 6, k = idx & 63;
  float v = 0.5f * (agg[(size_t)n * 128 + k] + agg[(size_t)n * 128 + 64 + k]) + bias[k];
  h16p[idx] = (_Float16)fmaxf(v, 0.f);
}

// ---------- layer-2 transform with WMMA: hpre2 = h1(f16) @ W2 -----------------
// One wave per 16-row tile; K=64 (2 k-steps), Nout=128 (8 n-tiles) -> 16 WMMAs.
__global__ void __launch_bounds__(256)
gemm2_wmma_kernel(const _Float16* __restrict__ A, const _Float16* __restrict__ Bp,
                  float* __restrict__ C, int nRows) {
  const int K = 64, NT = 8, KS = 2, NOUT = 128;
  int wid = (blockIdx.x * blockDim.x + threadIdx.x) >> 5;
  int lane = threadIdx.x & 31;
  int mtiles = (nRows + 15) >> 4;
  if (wid >= mtiles) return;
  int mbase = wid << 4;
  int m = mbase + (lane & 15);
  if (m >= nRows) m = nRows - 1;          // keep all lanes active (EXEC all-1s)
  int ka = (lane < 16) ? 0 : 16;
  f8 c[NT] = {};
  for (int s = 0; s < KS; s++) {
    const _Float16* ar = A + (size_t)m * K + 32 * s + ka;
    h8 s0 = *(const h8*)ar;
    h8 s1 = *(const h8*)(ar + 8);
    h16 a;
#pragma unroll
    for (int i = 0; i < 8; i++) { a[i] = s0[i]; a[i + 8] = s1[i]; }
#pragma unroll
    for (int nt = 0; nt < NT; nt++) {
      h16 b = *(const h16*)(Bp + ((size_t)(s * NT + nt) * 32 + lane) * 16);
      c[nt] = __builtin_amdgcn_wmma_f32_16x16x32_f16(false, a, false, b,
                                                     (short)0, c[nt], false, false);
    }
  }
  int rofs = (lane < 16) ? 0 : 8, nl = lane & 15;
#pragma unroll
  for (int nt = 0; nt < NT; nt++)
#pragma unroll
    for (int r = 0; r < 8; r++) {
      int row = mbase + rofs + r;
      if (row < nRows) C[(size_t)row * NOUT + nt * 16 + nl] = c[nt][r];
    }
}

// ---------- classifier: WMMA(feat @ Wcls1)+relu -> LDS -> 64x4 GEMM + flags ----
__global__ void __launch_bounds__(256)
classifier_kernel(const _Float16* __restrict__ h2, const int* __restrict__ batch,
                  const _Float16* __restrict__ ce16, const _Float16* __restrict__ Bp,
                  const float* __restrict__ bcls1, const float* __restrict__ Wcls2,
                  const float* __restrict__ bcls2, const float* __restrict__ x,
                  const float* __restrict__ Wf1, const float* __restrict__ bf1,
                  const float* __restrict__ Wf2, const float* __restrict__ bf2,
                  float* __restrict__ out, int nRows) {
  const int NT = 4, KS = 4;
  __shared__ float g[8][16][68];
  int w = threadIdx.x >> 5;
  int lane = threadIdx.x & 31;
  int wid = (blockIdx.x * blockDim.x + threadIdx.x) >> 5;
  int mtiles = (nRows + 15) >> 4;
  bool active = wid < mtiles;
  int mbase = wid << 4;
  if (active) {
    int m = mbase + (lane & 15);
    if (m >= nRows) m = nRows - 1;
    int bm = batch[m];
    int ka = (lane < 16) ? 0 : 16;
    f8 c[NT] = {};
    for (int s = 0; s < KS; s++) {
      // K=128 feat: k<64 from h2[m], k>=64 from climber-embed[batch[m]]
      const _Float16* base = (s < 2) ? (h2 + (size_t)m * 64)
                                     : (ce16 + (size_t)bm * 64 - 64);
      int kofs = 32 * s + ka;
      h8 s0 = *(const h8*)(base + kofs);
      h8 s1 = *(const h8*)(base + kofs + 8);
      h16 a;
#pragma unroll
      for (int i = 0; i < 8; i++) { a[i] = s0[i]; a[i + 8] = s1[i]; }
#pragma unroll
      for (int nt = 0; nt < NT; nt++) {
        h16 b = *(const h16*)(Bp + ((size_t)(s * NT + nt) * 32 + lane) * 16);
        c[nt] = __builtin_amdgcn_wmma_f32_16x16x32_f16(false, a, false, b,
                                                       (short)0, c[nt], false, false);
      }
    }
    int rofs = (lane < 16) ? 0 : 8, nl = lane & 15;
#pragma unroll
    for (int nt = 0; nt < NT; nt++)
#pragma unroll
      for (int r = 0; r < 8; r++) {
        int col = nt * 16 + nl;
        g[w][rofs + r][col] = fmaxf(c[nt][r] + bcls1[col], 0.f);
      }
  }
  __syncthreads();
  if (active) {
    int ml = lane & 15;
    int gm = mbase + ml;
    if (gm < nRows) {
      float fx = x[(size_t)gm * 8 + 6];
      float fy = x[(size_t)gm * 8 + 7];
      float tj[8];
#pragma unroll
      for (int j = 0; j < 8; j++)
        tj[j] = fmaxf(fx * Wf1[j] + fy * Wf1[8 + j] + bf1[j], 0.f);
      int o0 = lane >> 4;
#pragma unroll
      for (int oi = 0; oi < 2; oi++) {
        int o = o0 + oi * 2;
        float acc = bcls2[o];
#pragma unroll 8
        for (int k = 0; k < 64; k++) acc += g[w][ml][k] * Wcls2[k * 4 + o];
        float fo = bf2[o];
#pragma unroll
        for (int j = 0; j < 8; j++) fo += tj[j] * Wf2[j * 4 + o];
        out[(size_t)gm * 4 + o] = acc + fo;
      }
    }
  }
}

extern "C" void kernel_launch(void* const* d_in, const int* in_sizes, int n_in,
                              void* d_out, int out_size, void* d_ws, size_t ws_size,
                              hipStream_t stream) {
  const float* x      = (const float*)d_in[0];
  const int*   ei     = (const int*)d_in[1];
  const int*   batch  = (const int*)d_in[2];
  const float* climber= (const float*)d_in[3];
  const float* W1     = (const float*)d_in[4];
  const float* a1s    = (const float*)d_in[5];
  const float* a1d    = (const float*)d_in[6];
  const float* b1     = (const float*)d_in[7];
  const float* W2     = (const float*)d_in[8];
  const float* a2s    = (const float*)d_in[9];
  const float* a2d    = (const float*)d_in[10];
  const float* b2     = (const float*)d_in[11];
  const float* Wc     = (const float*)d_in[12];
  const float* bc     = (const float*)d_in[13];
  const float* Wcls1  = (const float*)d_in[14];
  const float* bcls1  = (const float*)d_in[15];
  const float* Wcls2  = (const float*)d_in[16];
  const float* bcls2  = (const float*)d_in[17];
  const float* Wf1    = (const float*)d_in[18];
  const float* bf1    = (const float*)d_in[19];
  const float* Wf2    = (const float*)d_in[20];
  const float* bf2    = (const float*)d_in[21];

  const int N  = in_sizes[0] / 8;
  const int E  = in_sizes[1] / 2;
  const int G  = in_sizes[3] / 4;
  const int EN = E + N;

  char* p = (char*)d_ws;
  auto carve = [&](size_t bytes) {
    char* r = p;
    p += (bytes + 255) & ~(size_t)255;
    return r;
  };
  float*     hpre    = (float*)carve((size_t)N * 128 * 4);
  float*     agg     = (float*)carve((size_t)N * 128 * 4);
  float*     als     = (float*)carve((size_t)N * 2 * 4);
  float*     ald     = (float*)carve((size_t)N * 2 * 4);
  unsigned*  emaxkey = (unsigned*)carve((size_t)N * 2 * 4);
  float*     denom   = (float*)carve((size_t)N * 2 * 4);
  float*     exbuf   = (float*)carve((size_t)EN * 2 * 4);
  _Float16*  hcur16  = (_Float16*)carve((size_t)N * 64 * 2);
  _Float16*  ce16    = (_Float16*)carve((size_t)G * 64 * 2);
  _Float16*  W2p     = (_Float16*)carve((size_t)2 * 8 * 512 * 2);
  _Float16*  Wc1p    = (_Float16*)carve((size_t)4 * 4 * 512 * 2);
  (void)ws_size; (void)n_in; (void)out_size;

  const int TB = 256;
  auto cdiv = [](long long a, long long b) { return (unsigned)((a + b - 1) / b); };
  int mtiles = (N + 15) / 16;

  // weight prep (tiny)
  pack_b_kernel<<<cdiv(2 * 8 * 512, TB), TB, 0, stream>>>(W2, W2p, 2, 8, 128);
  pack_b_kernel<<<cdiv(4 * 4 * 512, TB), TB, 0, stream>>>(Wcls1, Wc1p, 4, 4, 64);
  climber_kernel<<<cdiv((long long)G * 64, TB), TB, 0, stream>>>(climber, Wc, bc, ce16, G);

  // ---- layer 1 ----
  gemm1_kernel<<<cdiv((long long)N * 32, TB), TB, 0, stream>>>(x, W1, hpre, N);
  alpha_init_kernel<<<cdiv((long long)N * 2, TB), TB, 0, stream>>>(hpre, a1s, a1d, als, ald, emaxkey, denom, N);
  hipMemsetAsync(agg, 0, (size_t)N * 128 * 4, stream);
  edge_max_kernel<<<cdiv(EN, TB), TB, 0, stream>>>(als, ald, ei, emaxkey, E, EN);
  edge_exp_kernel<<<cdiv(EN, TB), TB, 0, stream>>>(als, ald, ei, emaxkey, denom, exbuf, E, EN);
  edge_msg_kernel<<<cdiv((long long)EN * 32, TB), TB, 0, stream>>>(hpre, exbuf, denom, ei, agg, E, EN);
  finalize_kernel<<<cdiv((long long)N * 64, TB), TB, 0, stream>>>(agg, b1, hcur16, N);

  // ---- layer 2 ----
  gemm2_wmma_kernel<<<cdiv(mtiles, 8), TB, 0, stream>>>(hcur16, W2p, hpre, N);
  alpha_init_kernel<<<cdiv((long long)N * 2, TB), TB, 0, stream>>>(hpre, a2s, a2d, als, ald, emaxkey, denom, N);
  hipMemsetAsync(agg, 0, (size_t)N * 128 * 4, stream);
  edge_max_kernel<<<cdiv(EN, TB), TB, 0, stream>>>(als, ald, ei, emaxkey, E, EN);
  edge_exp_kernel<<<cdiv(EN, TB), TB, 0, stream>>>(als, ald, ei, emaxkey, denom, exbuf, E, EN);
  edge_msg_kernel<<<cdiv((long long)EN * 32, TB), TB, 0, stream>>>(hpre, exbuf, denom, ei, agg, E, EN);
  finalize_kernel<<<cdiv((long long)N * 64, TB), TB, 0, stream>>>(agg, b2, hcur16, N);

  // ---- classifier heads ----
  classifier_kernel<<<cdiv(mtiles, 8), TB, 0, stream>>>(hcur16, batch, ce16, Wc1p,
                                                        bcls1, Wcls2, bcls2, x,
                                                        Wf1, bf1, Wf2, bf2,
                                                        (float*)d_out, N);
}
